// SingleAttention_22402549416144
// MI455X (gfx1250) — matrix operand order repaired and checked by reading, not compiled, not run on hardware
//
#include <hip/hip_runtime.h>

#define SEQ    4096
#define DMODEL 1024
#define DK     128
#define QSCALE 0.08838834764831845f   // 1/sqrt(128)
#define NEGV   (-1.0e10f)

typedef __bf16          v16bf __attribute__((ext_vector_type(16)));
typedef float           v8f   __attribute__((ext_vector_type(8)));
typedef unsigned short  v16us __attribute__((ext_vector_type(16)));
typedef unsigned short  v8us  __attribute__((ext_vector_type(8)));
typedef unsigned int    v8u   __attribute__((ext_vector_type(8)));
typedef unsigned int    v4u   __attribute__((ext_vector_type(4)));
typedef int             v8i   __attribute__((ext_vector_type(8)));
typedef int             v4i   __attribute__((ext_vector_type(4)));

#if __has_builtin(__builtin_amdgcn_tensor_load_to_lds) && \
    __has_builtin(__builtin_amdgcn_s_wait_tensorcnt)
#define HAVE_TDM 1
#else
#define HAVE_TDM 0
#endif

// ---- cheap conversions -----------------------------------------------------
// round-half-up f32->bf16 (2 VALU)
__device__ __forceinline__ unsigned short f2bf(float f) {
  return (unsigned short)((__float_as_uint(f) + 0x8000u) >> 16);
}
// two f32 -> packed bf16x2 via v_perm_b32 (3 VALU for 2 values)
__device__ __forceinline__ unsigned int pack2bf(float a, float b) {
  unsigned ua = __float_as_uint(a) + 0x8000u;
  unsigned ub = __float_as_uint(b) + 0x8000u;
#if __has_builtin(__builtin_amdgcn_perm)
  return __builtin_amdgcn_perm(ub, ua, 0x07060302u);  // {ub[31:16], ua[31:16]}
#else
  return (ua >> 16) | (ub & 0xFFFF0000u);
#endif
}

__device__ __forceinline__ v8f wmma_bf16(v16us a, v16us b, v8f c) {
  union { v16us u; v16bf v; } A, B;
  A.u = a; B.u = b;
  return __builtin_amdgcn_wmma_f32_16x16x32_bf16(
      false, A.v, false, B.v, (short)0, c, false, false);
}

// ---- Tensor Data Mover: 2D tile -> LDS (D# per CDNA5 ISA sec.8) ---------
#if HAVE_TDM
__device__ __forceinline__ unsigned ldsoff(const void* p) {
  return (unsigned)(unsigned long long)(uintptr_t)p;  // low 32b = LDS offset
}
__device__ __forceinline__ void tdm_load_2d(unsigned lds_addr, const void* gptr,
                                            unsigned tile_d0, unsigned tile_d1,
                                            unsigned tensor_d0, unsigned stride0,
                                            unsigned tensor_d1) {
  unsigned long long ga = (unsigned long long)(uintptr_t)gptr;
  v4u g0;
  g0[0] = 1u;                                    // count=1 (valid), user mode
  g0[1] = lds_addr;                              // lds_addr (bytes)
  g0[2] = (unsigned)ga;                          // global_addr[31:0]
  g0[3] = (unsigned)(ga >> 32) | (2u << 30);     // global_addr[56:32] | type=2
  v8i g1;
  g1[0] = (int)(1u << 16);                       // wg_mask=0, data_size=1 (2B)
  g1[1] = (int)((tensor_d0 & 0xFFFFu) << 16);    // tensor_dim0[15:0]
  g1[2] = (int)(((tensor_d0 >> 16) & 0xFFFFu) |
                ((tensor_d1 & 0xFFFFu) << 16));  // td0[31:16] | td1[15:0]
  g1[3] = (int)(((tensor_d1 >> 16) & 0xFFFFu) |
                ((tile_d0 & 0xFFFFu) << 16));    // td1[31:16] | tile_dim0
  g1[4] = (int)(tile_d1 & 0xFFFFu);              // tile_dim1 (tile_dim2=0)
  g1[5] = (int)stride0;                          // tensor_dim0_stride[31:0]
  g1[6] = 0;                                     // stride0[47:32] | d1_stride lo
  g1[7] = 0;
  v4i g2 = {0, 0, 0, 0};
  v4i g3 = {0, 0, 0, 0};
#if defined(__clang_major__) && __clang_major__ >= 23
  v8i gx = {0, 0, 0, 0, 0, 0, 0, 0};
  __builtin_amdgcn_tensor_load_to_lds(g0, g1, g2, g3, gx, 0);
#else
  __builtin_amdgcn_tensor_load_to_lds(g0, g1, g2, g3, 0);
#endif
}
#endif  // HAVE_TDM

// ---------------------------------------------------------------------------
// Kernel 0: W [DMODEL][DK] f32 -> WT [DK][DMODEL] bf16 (B-operand friendly)
// ---------------------------------------------------------------------------
__global__ void wconv_kernel(const float* __restrict__ Wq,
                             const float* __restrict__ Wk,
                             const float* __restrict__ Wv,
                             unsigned short* __restrict__ wt) {
  const float* W = (blockIdx.y == 0) ? Wq : (blockIdx.y == 1) ? Wk : Wv;
  unsigned short* o = wt + (size_t)blockIdx.y * DK * DMODEL;
  const int n = blockIdx.x;
  for (int k = threadIdx.x; k < DMODEL; k += blockDim.x)
    o[n * DMODEL + k] = f2bf(W[k * DK + n]);
}

// ---------------------------------------------------------------------------
// Kernel 1: projections via bf16 WMMA (f32 accumulate), bias in C-init.
// ---------------------------------------------------------------------------
__global__ __launch_bounds__(128) void proj_kernel(
    const float* __restrict__ q, const float* __restrict__ k,
    const float* __restrict__ v, const unsigned short* __restrict__ wt,
    const float* __restrict__ bq, const float* __restrict__ bk,
    const float* __restrict__ bv,
    unsigned short* __restrict__ qp, unsigned short* __restrict__ kp,
    unsigned short* __restrict__ vpT) {
  const int wid  = blockIdx.x * (blockDim.x >> 5) + (threadIdx.x >> 5);
  const int mat  = wid >> 11;
  const int rem  = wid & 2047;
  const int mt   = rem >> 3;
  const int nt   = rem & 7;
  const int lane = threadIdx.x & 31;
  const int ln   = lane & 15;
  const int hl   = lane >> 4;

  const float* x    = (mat == 0) ? q  : (mat == 1) ? k  : v;
  const float* bias = (mat == 0) ? bq : (mat == 1) ? bk : bv;
  const unsigned short* wtm = wt + (size_t)mat * DK * DMODEL;

  const int row = mt * 16 + ln;
  const int col = nt * 16 + ln;

  const float bval = bias[col];
  v8f acc = {bval, bval, bval, bval, bval, bval, bval, bval};

  const float* xrow = x + (size_t)row * DMODEL;
  const unsigned short* wcol = wtm + (size_t)col * DMODEL + 16 * hl;

  for (int ks = 0; ks < DMODEL; ks += 32) {
    v8f a0 = *(const v8f*)(xrow + ks + 16 * hl);       // K = ks+16*hl+0..7
    v8f a1 = *(const v8f*)(xrow + ks + 16 * hl + 8);   // K = ks+16*hl+8..15
    union { v8u u; v16us s; } au;
#pragma unroll
    for (int e = 0; e < 4; ++e) {
      au.u[e]     = pack2bf(a0[2 * e], a0[2 * e + 1]);
      au.u[4 + e] = pack2bf(a1[2 * e], a1[2 * e + 1]);
    }
    v16us bu = *(const v16us*)(wcol + ks);
    acc = wmma_bf16(au.s, bu, acc);
  }

  if (mat == 2) {  // vpT: 8 consecutive rows per lane -> one packed b128 store
    union { v4u u; v8us s; } pk;
#pragma unroll
    for (int e = 0; e < 4; ++e) pk.u[e] = pack2bf(acc[2 * e], acc[2 * e + 1]);
    *(v8us*)(vpT + (size_t)col * SEQ + mt * 16 + 8 * hl) = pk.s;
  } else {
    unsigned short* dst = (mat == 0) ? qp : kp;
    const float scale = (mat == 0) ? QSCALE : 1.0f;
#pragma unroll
    for (int r = 0; r < 8; ++r)
      dst[(size_t)(mt * 16 + r + 8 * hl) * DK + col] = f2bf(acc[r] * scale);
  }
}

// ---------------------------------------------------------------------------
// Kernel 2: causal flash attention, one wave per 16-query block.
// K/V tiles double-buffered in LDS via the Tensor Data Mover; DMA for pair
// jp+1 overlaps compute of pair jp (s_wait_tensorcnt <= 2).
// ---------------------------------------------------------------------------
__global__ __launch_bounds__(32) void attn_kernel(
    const unsigned short* __restrict__ qp, const unsigned short* __restrict__ kp,
    const unsigned short* __restrict__ vpT, float* __restrict__ out) {
#if HAVE_TDM
  __shared__ unsigned short kbuf[2][32 * DK];   // [key local 0..31][d 0..127]
  __shared__ unsigned short vbuf[2][DK * 32];   // [d 0..127][key local 0..31]
#endif
  __shared__ unsigned short p_lds[16 * 32];

  const int qb   = blockIdx.x;
  const int lane = threadIdx.x & 31;
  const int ln   = lane & 15;
  const int hl   = lane >> 4;

  // Q block: 4 resident A-fragments (K = 128 in steps of 32)
  v16us qa[4];
  {
    const unsigned short* qrow = qp + (size_t)(qb * 16 + ln) * DK;
#pragma unroll
    for (int d = 0; d < 4; ++d) {
      union { v8us h[2]; v16us v; } t;
      t.h[0] = *(const v8us*)(qrow + d * 32 + 16 * hl);
      t.h[1] = *(const v8us*)(qrow + d * 32 + 16 * hl + 8);
      qa[d] = t.v;
    }
  }

  v8f o[8];
#pragma unroll
  for (int n = 0; n < 8; ++n) o[n] = (v8f)(0.0f);
  float mrow[8], lrow[8];
#pragma unroll
  for (int r = 0; r < 8; ++r) { mrow[r] = -3.0e38f; lrow[r] = 0.0f; }

  const int ntk    = qb + 1;
  const int npairs = (ntk + 1) >> 1;

#if HAVE_TDM
  // prologue: DMA pair 0 into buffer 0
  tdm_load_2d(ldsoff(&kbuf[0][0]), kp, DK, 32, DK, DK, SEQ);
  tdm_load_2d(ldsoff(&vbuf[0][0]), vpT, 32, DK, SEQ, SEQ, DK);
#endif

  for (int jp = 0; jp < npairs; ++jp) {
#if HAVE_TDM
    if (jp + 1 < npairs) {  // DMA next pair while computing this one
      const int b = (jp + 1) & 1;
      tdm_load_2d(ldsoff(&kbuf[b][0]), kp + (size_t)(jp + 1) * 32 * DK,
                  DK, 32, DK, DK, SEQ);
      tdm_load_2d(ldsoff(&vbuf[b][0]), vpT + (size_t)(jp + 1) * 32,
                  32, DK, SEQ, SEQ, DK);
      __builtin_amdgcn_s_wait_tensorcnt((short)2);  // current pair landed
    } else {
      __builtin_amdgcn_s_wait_tensorcnt((short)0);
    }
    asm volatile("" ::: "memory");
    const unsigned short* kt = &kbuf[jp & 1][0];
    const unsigned short* vt = &vbuf[jp & 1][0];
#else
    if (2 * jp + 2 <= qb)
      __builtin_prefetch(kp + (size_t)((2 * jp + 2) * 16 + ln) * DK, 0, 0);
#endif

    // ---- scores: two 16x16 tiles (keys 32*jp .. 32*jp+31) ----
    v8f s[2];
#pragma unroll
    for (int t = 0; t < 2; ++t) {
      const int j = 2 * jp + t;
      s[t] = (v8f)(0.0f);
      if (j <= qb) {
#if HAVE_TDM
        const unsigned short* krow = kt + (size_t)(t * 16 + ln) * DK + 16 * hl;
#else
        const unsigned short* krow = kp + (size_t)(j * 16 + ln) * DK + 16 * hl;
#endif
#pragma unroll
        for (int d = 0; d < 4; ++d) {
          v16us kb = *(const v16us*)(krow + d * 32);
          s[t] = wmma_bf16(qa[d], kb, s[t]);
        }
        if (j == qb) {
#pragma unroll
          for (int r = 0; r < 8; ++r)
            if (ln > r + 8 * hl) s[t][r] = NEGV;
        }
      } else {
#pragma unroll
        for (int r = 0; r < 8; ++r) s[t][r] = NEGV;
      }
    }

    // ---- online softmax ----
    float pr0[8], pr1[8];
#pragma unroll
    for (int r = 0; r < 8; ++r) {
      float rm = fmaxf(s[0][r], s[1][r]);
      rm = fmaxf(rm, __shfl_xor(rm, 1, 32));
      rm = fmaxf(rm, __shfl_xor(rm, 2, 32));
      rm = fmaxf(rm, __shfl_xor(rm, 4, 32));
      rm = fmaxf(rm, __shfl_xor(rm, 8, 32));
      const float mnew = fmaxf(mrow[r], rm);
      const float corr = __expf(mrow[r] - mnew);
      mrow[r] = mnew;
      const float p0 = __expf(s[0][r] - mnew);
      const float p1 = __expf(s[1][r] - mnew);
      pr0[r] = p0; pr1[r] = p1;
      float ps = p0 + p1;
      ps += __shfl_xor(ps, 1, 32);
      ps += __shfl_xor(ps, 2, 32);
      ps += __shfl_xor(ps, 4, 32);
      ps += __shfl_xor(ps, 8, 32);
      lrow[r] = lrow[r] * corr + ps;
#pragma unroll
      for (int n = 0; n < 8; ++n) o[n][r] *= corr;
    }

    // ---- C-layout P -> row-major LDS -> A-layout fragment ----
#pragma unroll
    for (int r = 0; r < 8; ++r) {
      p_lds[(r + 8 * hl) * 32 + ln]      = f2bf(pr0[r]);
      p_lds[(r + 8 * hl) * 32 + 16 + ln] = f2bf(pr1[r]);
    }
    __syncthreads();
    v16us pa;
    {
      union { v8us h[2]; v16us v; } t;
      const unsigned short* pp = p_lds + ln * 32 + 16 * hl;
      t.h[0] = *(const v8us*)pp;
      t.h[1] = *(const v8us*)(pp + 8);
      pa = t.v;
    }

    // ---- O += P @ V ----
#pragma unroll
    for (int n = 0; n < 8; ++n) {
#if HAVE_TDM
      v16us vb = *(const v16us*)(vt + (size_t)(n * 16 + ln) * 32 + 16 * hl);
#else
      v16us vb = *(const v16us*)(vpT + (size_t)(n * 16 + ln) * SEQ +
                                 jp * 32 + 16 * hl);
#endif
      o[n] = wmma_bf16(pa, vb, o[n]);
    }
    __syncthreads();
  }

  // ---- epilogue ----
  const int orow0 = qb * 16 + 8 * hl;
#pragma unroll
  for (int n = 0; n < 8; ++n)
#pragma unroll
    for (int r = 0; r < 8; ++r)
      out[(size_t)(orow0 + r) * DK + n * 16 + ln] = o[n][r] / lrow[r];
}

// ---------------------------------------------------------------------------
extern "C" void kernel_launch(void* const* d_in, const int* in_sizes, int n_in,
                              void* d_out, int out_size, void* d_ws, size_t ws_size,
                              hipStream_t stream) {
  const float* q  = (const float*)d_in[0];
  const float* k  = (const float*)d_in[1];
  const float* v  = (const float*)d_in[2];
  const float* Wq = (const float*)d_in[3];
  const float* bq = (const float*)d_in[4];
  const float* Wk = (const float*)d_in[5];
  const float* bk = (const float*)d_in[6];
  const float* Wv = (const float*)d_in[7];
  const float* bv = (const float*)d_in[8];

  unsigned short* ws  = (unsigned short*)d_ws;
  unsigned short* wt  = ws;                          // 3 * 128*1024 bf16
  unsigned short* qp  = wt + 3 * DK * DMODEL;        // 4096*128 bf16
  unsigned short* kp  = qp + SEQ * DK;               // 4096*128 bf16
  unsigned short* vpT = kp + SEQ * DK;               // 128*4096 bf16 (transposed)

  wconv_kernel<<<dim3(DK, 3), 256, 0, stream>>>(Wq, Wk, Wv, wt);
  proj_kernel<<<dim3(3 * 2048 / 4), 128, 0, stream>>>(q, k, v, wt, bq, bk, bv,
                                                      qp, kp, vpT);
  attn_kernel<<<dim3(SEQ / 16), 32, 0, stream>>>(qp, kp, vpT, (float*)d_out);
}
